// VisionAttention_29738353557811
// MI455X (gfx1250) — matrix operand order repaired and checked by reading, not compiled, not run on hardware
//
#include <hip/hip_runtime.h>

#define N_TOK 16384
#define DMODEL 1280
#define NHEAD 16
#define HDIM 80
#define HPAD 96
#define SEGLEN 2048
#define NSEGS 8

typedef unsigned short bf16_t;
typedef __attribute__((ext_vector_type(16))) __bf16 v16bf;
typedef __attribute__((ext_vector_type(8)))  float  v8f;

union Frag {
    v16bf v;
    unsigned u[8];
};

__device__ __forceinline__ bf16_t f2b(float f) {
    union { float f; unsigned u; } x; x.f = f;
    unsigned r = x.u + 0x7FFFu + ((x.u >> 16) & 1u);  // round-to-nearest-even
    return (bf16_t)(r >> 16);
}
__device__ __forceinline__ float b2f(bf16_t b) {
    union { unsigned u; float f; } x; x.u = ((unsigned)b) << 16;
    return x.f;
}

// CDNA5 async global->LDS copy, 16 bytes per active lane. Tracked by ASYNCcnt.
// VDST = raw LDS byte address (low 32 bits of the generic pointer), VADDR = 64b
// global address. Completion enforced with s_wait_asynccnt.
__device__ __forceinline__ void async_load16(unsigned lds_addr, const void* gaddr) {
    asm volatile("global_load_async_to_lds_b128 %0, %1, off"
                 :: "v"(lds_addr), "v"(gaddr) : "memory");
}

// A-matrix 16x32 bf16 fragment (ISA 7.12.2): lane<16 row=lane, K={0..7,16..23};
// lane>=16 row=lane-16, K={8..15,24..31}. base points at element [0][0] of the
// 16x32 tile in a row-major bf16 buffer with row stride ld (even).
__device__ __forceinline__ v16bf load_a_frag(const bf16_t* base, int ld, int lane) {
    Frag f;
    const int half = lane >> 4;
    const unsigned* row = (const unsigned*)(base + (lane & 15) * ld);
#pragma unroll
    for (int i = 0; i < 4; ++i) {
        f.u[i]     = row[half * 4 + i];       // K pairs 0..3 / 4..7
        f.u[4 + i] = row[8 + half * 4 + i];   // K pairs 8..11 / 12..15
    }
    return f.v;
}

// B-matrix 32x16 bf16 fragment from a "Bt" buffer: row index = output column n,
// contiguous along K. lane<16 holds K={0..7,16..23}, lane>=16 K={8..15,24..31}.
__device__ __forceinline__ v16bf load_b_frag(const bf16_t* base, int ld, int lane) {
    Frag f;
    const int half = lane >> 4;
    const unsigned* row = (const unsigned*)(base + (lane & 15) * ld);
#pragma unroll
    for (int i = 0; i < 8; ++i) f.u[i] = row[(i >> 2) * 8 + half * 4 + (i & 3)];
    return f.v;
}

__device__ __forceinline__ v8f wmma_bf16(v16bf a, v16bf b, v8f c) {
    return __builtin_amdgcn_wmma_f32_16x16x32_bf16(false, a, false, b, (short)0, c,
                                                   false, false);
}

// ---------------- elementwise conversion kernels ----------------

__global__ __launch_bounds__(256) void cvt_f32_bf16(const float* __restrict__ src,
                                                    bf16_t* __restrict__ dst, size_t n) {
    size_t i = (size_t)blockIdx.x * blockDim.x + threadIdx.x;
    if (i < n) dst[i] = f2b(src[i]);
}

// src [rows][cols] f32 -> dst [cols][rows] bf16 (weight transpose so GEMM B
// tiles are contiguous along the contraction dim)
__global__ __launch_bounds__(256) void cvt_f32_bf16_t(const float* __restrict__ src,
                                                      bf16_t* __restrict__ dst,
                                                      int rows, int cols) {
    size_t i = (size_t)blockIdx.x * blockDim.x + threadIdx.x;
    if (i >= (size_t)rows * cols) return;
    int c = (int)(i / rows);
    int r = (int)(i % rows);
    dst[(size_t)c * rows + r] = f2b(src[(size_t)r * cols + c]);
}

// ---------------- WMMA GEMM: C[M][Nout] = A[M][Kd] * Bt[Nout][Kd]^T + bias ----
// Async-double-buffered: each thread issues exactly 3 async b128 copies per
// tile, so "s_wait_asynccnt 3" after issuing the next tile drains the previous
// one (async loads complete in order).

template <bool OUT_BF16>
__global__ __launch_bounds__(256) void gemm_bf16_wmma(
    const bf16_t* __restrict__ A, const bf16_t* __restrict__ Bt,
    const float* __restrict__ bias, void* __restrict__ C,
    int M, int Nout, int Kd) {
    constexpr int BM = 128, BN = 64, BK = 32;
    constexpr int LDA = BK + 16;  // 48 elems -> 96B row stride (16B aligned)
    constexpr int LDB = BK + 16;
    __shared__ bf16_t sA[2][BM * LDA];
    __shared__ bf16_t sB[2][BN * LDB];

    const int tid  = threadIdx.x;
    const int lane = tid & 31;
    const int wave = tid >> 5;
    const int half = lane >> 4;
    const int mBase = blockIdx.y * BM;
    const int nBase = blockIdx.x * BN;
    const int wm = (wave & 3) * 32;   // wave tile: 32x32
    const int wn = (wave >> 2) * 32;

    auto issueTile = [&](int kt, int buf) {
        {   // A tile: 128 rows x 32 bf16, 2 threads/row, 32B each (2 asyncs)
            int row = tid >> 1, sgm = tid & 1;
            const char* g = (const char*)(A + (size_t)(mBase + row) * Kd + kt + sgm * 16);
            unsigned l = (unsigned)(size_t)(&sA[buf][row * LDA + sgm * 16]);
            async_load16(l, g);
            async_load16(l + 16, g + 16);
        }
        {   // B tile: 64 rows x 32 bf16, 4 threads/row, 16B each (1 async)
            int row = tid >> 2, sgm = tid & 3;
            const char* g = (const char*)(Bt + (size_t)(nBase + row) * Kd + kt + sgm * 8);
            unsigned l = (unsigned)(size_t)(&sB[buf][row * LDB + sgm * 8]);
            async_load16(l, g);
        }
    };

    const v8f vz = {0.f, 0.f, 0.f, 0.f, 0.f, 0.f, 0.f, 0.f};
    v8f acc[2][2] = {{vz, vz}, {vz, vz}};

    const int NK = Kd / BK;
    issueTile(0, 0);
    for (int t = 0; t < NK; ++t) {
        const int buf = t & 1;
        if (t + 1 < NK) {
            issueTile((t + 1) * BK, buf ^ 1);
            asm volatile("s_wait_asynccnt 0x3" ::: "memory");  // prev tile landed
        } else {
            asm volatile("s_wait_asynccnt 0x0" ::: "memory");
        }
        __syncthreads();

        v16bf a0 = load_a_frag(&sA[buf][(wm + 0)  * LDA], LDA, lane);
        v16bf a1 = load_a_frag(&sA[buf][(wm + 16) * LDA], LDA, lane);
        v16bf b0 = load_b_frag(&sB[buf][(wn + 0)  * LDB], LDB, lane);
        v16bf b1 = load_b_frag(&sB[buf][(wn + 16) * LDB], LDB, lane);
        acc[0][0] = wmma_bf16(a0, b0, acc[0][0]);
        acc[0][1] = wmma_bf16(a0, b1, acc[0][1]);
        acc[1][0] = wmma_bf16(a1, b0, acc[1][0]);
        acc[1][1] = wmma_bf16(a1, b1, acc[1][1]);
        __syncthreads();   // reads done before this buffer is refilled
    }

#pragma unroll
    for (int mi = 0; mi < 2; ++mi)
#pragma unroll
        for (int ni = 0; ni < 2; ++ni) {
            int col = nBase + wn + ni * 16 + (lane & 15);
            float bv = bias[col];
#pragma unroll
            for (int r = 0; r < 8; ++r) {
                int row = mBase + wm + mi * 16 + (half ? 8 + r : r);
                float v = acc[mi][ni][r] + bv;
                if (OUT_BF16)
                    ((bf16_t*)C)[(size_t)row * Nout + col] = f2b(v);
                else
                    ((float*)C)[(size_t)row * Nout + col] = v;
            }
        }
}

// ---------------- RoPE + scatter to padded/attention-friendly layouts --------
// qkv [N][3*1280] bf16 -> qh,kh [H][N][96] bf16 (zero-padded), vt [H][96][N]

__global__ __launch_bounds__(256) void rope_scatter(
    const bf16_t* __restrict__ qkv, const float* __restrict__ cosNK,
    const float* __restrict__ sinNK, bf16_t* __restrict__ qh,
    bf16_t* __restrict__ kh, bf16_t* __restrict__ vt) {
    size_t i = (size_t)blockIdx.x * blockDim.x + threadIdx.x;
    if (i >= (size_t)N_TOK * NHEAD * HPAD) return;
    int kp = (int)(i % HPAD);
    int nh = (int)(i / HPAD);
    int h  = nh % NHEAD;
    int n  = nh / NHEAD;

    float qv = 0.f, kv = 0.f, vv = 0.f;
    if (kp < HDIM) {
        size_t rb = (size_t)n * (3 * DMODEL);
        float q = b2f(qkv[rb + h * HDIM + kp]);
        float k = b2f(qkv[rb + DMODEL + h * HDIM + kp]);
        vv      = b2f(qkv[rb + 2 * DMODEL + h * HDIM + kp]);
        int  kr  = (kp < HDIM / 2) ? kp + HDIM / 2 : kp - HDIM / 2;
        float sg = (kp < HDIM / 2) ? -1.f : 1.f;
        float qr = sg * b2f(qkv[rb + h * HDIM + kr]);
        float kk = sg * b2f(qkv[rb + DMODEL + h * HDIM + kr]);
        float c = cosNK[(size_t)n * HDIM + kp];
        float s = sinNK[(size_t)n * HDIM + kp];
        qv = q * c + qr * s;
        kv = k * c + kk * s;
    }
    size_t o = ((size_t)h * N_TOK + n) * HPAD + kp;
    qh[o] = f2b(qv);
    kh[o] = f2b(kv);
    vt[((size_t)h * HPAD + kp) * N_TOK + n] = f2b(vv);
}

// ---------------- flash attention over one (segment, head, 128-query block) --
// Async-double-buffered K/V tiles: each thread issues exactly 6 async b128
// copies per key block (K: 64x96, V: 96x64 incl. zero pad rows 80..95).

__global__ __launch_bounds__(256) void attn_wmma(
    const bf16_t* __restrict__ qh, const bf16_t* __restrict__ kh,
    const bf16_t* __restrict__ vt, bf16_t* __restrict__ outb) {
    constexpr int BQ = 128, BKV = 64;
    constexpr int LDK = HPAD + 16;  // 112 -> 224B stride
    constexpr int LDV = BKV + 16;   // 80  -> 160B stride
    constexpr int LDP = BKV + 16;   // 80
    __shared__ bf16_t sK[2][BKV * LDK];    // 64 keys x 96 feat
    __shared__ bf16_t sV[2][HPAD * LDV];   // 96 feat x 64 keys (Vt, rows 80+ zero)
    __shared__ bf16_t sP[8 * 16 * LDP];    // per-wave P staging

    const int tid = threadIdx.x, lane = tid & 31, wave = tid >> 5, half = lane >> 4;
    const int head = blockIdx.y;
    const int seg  = blockIdx.z;
    const int segStart = seg * SEGLEN;
    const int qBase = segStart + blockIdx.x * BQ;
    const float scale = 0.1118033988749895f;  // 1/sqrt(80)

    auto issueKV = [&](int kb, int buf) {
        // K tile: 768 x 16B chunks, 3 per thread
        for (int i = tid; i < BKV * 12; i += 256) {
            int row = i / 12, sgm = i % 12;
            const char* g = (const char*)(kh + ((size_t)head * N_TOK + segStart + kb + row) * HPAD + sgm * 8);
            async_load16((unsigned)(size_t)(&sK[buf][row * LDK + sgm * 8]), g);
        }
        // Vt tile: 768 x 16B chunks, 3 per thread
        for (int i = tid; i < HPAD * 8; i += 256) {
            int row = i / 8, sgm = i % 8;
            const char* g = (const char*)(vt + ((size_t)head * HPAD + row) * N_TOK + segStart + kb + sgm * 8);
            async_load16((unsigned)(size_t)(&sV[buf][row * LDV + sgm * 8]), g);
        }
    };

    // Q fragments for this wave's 16 rows (96 = 3 K-chunks), kept in registers
    const bf16_t* qrow = qh + ((size_t)head * N_TOK + qBase + wave * 16) * HPAD;
    v16bf aq[3];
#pragma unroll
    for (int c3 = 0; c3 < 3; ++c3) aq[c3] = load_a_frag(qrow + c3 * 32, HPAD, lane);

    const v8f vz = {0.f, 0.f, 0.f, 0.f, 0.f, 0.f, 0.f, 0.f};
    float m[8], l[8];
    v8f o[5];
#pragma unroll
    for (int r = 0; r < 8; ++r) { m[r] = -1e30f; l[r] = 0.f; }
#pragma unroll
    for (int ft = 0; ft < 5; ++ft) o[ft] = vz;

    constexpr int NKB = SEGLEN / BKV;
    issueKV(0, 0);
    for (int t = 0; t < NKB; ++t) {
        const int buf = t & 1;
        if (t + 1 < NKB) {
            issueKV((t + 1) * BKV, buf ^ 1);
            asm volatile("s_wait_asynccnt 0x6" ::: "memory");  // prev tiles landed
        } else {
            asm volatile("s_wait_asynccnt 0x0" ::: "memory");
        }
        __syncthreads();

        // S = Q * K^T (16 x 64 per wave), f32 accum
        v8f s[4];
#pragma unroll
        for (int nt = 0; nt < 4; ++nt) {
            v8f c = vz;
#pragma unroll
            for (int c3 = 0; c3 < 3; ++c3) {
                v16bf bk = load_b_frag(&sK[buf][(nt * 16) * LDK + c3 * 32], LDK, lane);
                c = wmma_bf16(aq[c3], bk, c);
            }
            s[nt] = c * scale;
        }

        // online softmax (rows live in 16-lane groups; reduce with xor 1,2,4,8)
        float mnew[8];
#pragma unroll
        for (int r = 0; r < 8; ++r) {
            mnew[r] = m[r];
#pragma unroll
            for (int nt = 0; nt < 4; ++nt) mnew[r] = fmaxf(mnew[r], s[nt][r]);
        }
#pragma unroll
        for (int d = 1; d < 16; d <<= 1)
#pragma unroll
            for (int r = 0; r < 8; ++r) mnew[r] = fmaxf(mnew[r], __shfl_xor(mnew[r], d, 32));

        float alpha[8], rowsum[8];
#pragma unroll
        for (int r = 0; r < 8; ++r) {
            alpha[r] = __expf(m[r] - mnew[r]);
            m[r] = mnew[r];
            rowsum[r] = 0.f;
        }
#pragma unroll
        for (int nt = 0; nt < 4; ++nt)
#pragma unroll
            for (int r = 0; r < 8; ++r) {
                float p = __expf(s[nt][r] - mnew[r]);
                s[nt][r] = p;
                rowsum[r] += p;
            }
#pragma unroll
        for (int d = 1; d < 16; d <<= 1)
#pragma unroll
            for (int r = 0; r < 8; ++r) rowsum[r] += __shfl_xor(rowsum[r], d, 32);
#pragma unroll
        for (int r = 0; r < 8; ++r) l[r] = l[r] * alpha[r] + rowsum[r];
#pragma unroll
        for (int ft = 0; ft < 5; ++ft)
#pragma unroll
            for (int r = 0; r < 8; ++r) o[ft][r] *= alpha[r];

        // stage P (16x64 bf16) in per-wave LDS to re-fragment as A-matrix
        bf16_t* pw = sP + wave * 16 * LDP;
#pragma unroll
        for (int nt = 0; nt < 4; ++nt)
#pragma unroll
            for (int r = 0; r < 8; ++r)
                pw[(half ? 8 + r : r) * LDP + nt * 16 + (lane & 15)] = f2b(s[nt][r]);
        asm volatile("s_wait_dscnt 0" ::: "memory");  // same-wave LDS RAW

        // O += P * V
#pragma unroll
        for (int ft = 0; ft < 5; ++ft) {
            v8f c = o[ft];
#pragma unroll
            for (int kc = 0; kc < 2; ++kc) {
                v16bf ap = load_a_frag(pw + kc * 32, LDP, lane);
                v16bf bv = load_b_frag(&sV[buf][(ft * 16) * LDV + kc * 32], LDV, lane);
                c = wmma_bf16(ap, bv, c);
            }
            o[ft] = c;
        }
        __syncthreads();   // LDS reads done before this buffer is refilled
    }

    float inv[8];
#pragma unroll
    for (int r = 0; r < 8; ++r) inv[r] = 1.0f / l[r];
#pragma unroll
    for (int ft = 0; ft < 5; ++ft) {
        int f = ft * 16 + (lane & 15);
#pragma unroll
        for (int r = 0; r < 8; ++r) {
            int row = qBase + wave * 16 + (half ? 8 + r : r);
            outb[(size_t)row * DMODEL + head * HDIM + f] = f2b(o[ft][r] * inv[r]);
        }
    }
}

// ---------------- host launcher ----------------

extern "C" void kernel_launch(void* const* d_in, const int* in_sizes, int n_in,
                              void* d_out, int out_size, void* d_ws, size_t ws_size,
                              hipStream_t stream) {
    const float* hidden = (const float*)d_in[0];
    const float* cosNK  = (const float*)d_in[1];
    const float* sinNK  = (const float*)d_in[2];
    const float* qkv_w  = (const float*)d_in[3];
    const float* qkv_b  = (const float*)d_in[4];
    const float* proj_w = (const float*)d_in[5];
    const float* proj_b = (const float*)d_in[6];
    float* out = (float*)d_out;
    (void)in_sizes; (void)n_in; (void)out_size; (void)ws_size;

    char* ws = (char*)d_ws;
    size_t off = 0;
    auto salloc = [&](size_t bytes) -> void* {
        void* p = ws + off;
        off = (off + bytes + 255) & ~(size_t)255;
        return p;
    };
    bf16_t* hidden_b = (bf16_t*)salloc((size_t)N_TOK * DMODEL * 2);
    bf16_t* qkv_wt   = (bf16_t*)salloc((size_t)3 * DMODEL * DMODEL * 2);
    bf16_t* proj_wt  = (bf16_t*)salloc((size_t)DMODEL * DMODEL * 2);
    bf16_t* qkv_b16  = (bf16_t*)salloc((size_t)N_TOK * 3 * DMODEL * 2);
    bf16_t* qh       = (bf16_t*)salloc((size_t)NHEAD * N_TOK * HPAD * 2);
    bf16_t* khp      = (bf16_t*)salloc((size_t)NHEAD * N_TOK * HPAD * 2);
    bf16_t* vtp      = (bf16_t*)salloc((size_t)NHEAD * HPAD * N_TOK * 2);
    bf16_t* attn_b   = (bf16_t*)salloc((size_t)N_TOK * DMODEL * 2);

    // 1) conversions
    {
        size_t n = (size_t)N_TOK * DMODEL;
        cvt_f32_bf16<<<(unsigned)((n + 255) / 256), 256, 0, stream>>>(hidden, hidden_b, n);
    }
    {
        size_t n = (size_t)DMODEL * 3 * DMODEL;
        cvt_f32_bf16_t<<<(unsigned)((n + 255) / 256), 256, 0, stream>>>(qkv_w, qkv_wt, DMODEL, 3 * DMODEL);
    }
    {
        size_t n = (size_t)DMODEL * DMODEL;
        cvt_f32_bf16_t<<<(unsigned)((n + 255) / 256), 256, 0, stream>>>(proj_w, proj_wt, DMODEL, DMODEL);
    }
    // 2) QKV GEMM -> bf16 qkv
    gemm_bf16_wmma<true><<<dim3(3 * DMODEL / 64, N_TOK / 128), 256, 0, stream>>>(
        hidden_b, qkv_wt, qkv_b, (void*)qkv_b16, N_TOK, 3 * DMODEL, DMODEL);
    // 3) RoPE + scatter
    {
        size_t n = (size_t)N_TOK * NHEAD * HPAD;
        rope_scatter<<<(unsigned)((n + 255) / 256), 256, 0, stream>>>(qkv_b16, cosNK, sinNK, qh, khp, vtp);
    }
    // 4) segmented attention
    attn_wmma<<<dim3(SEGLEN / 128, NHEAD, NSEGS), 256, 0, stream>>>(qh, khp, vtp, attn_b);
    // 5) output projection -> f32 out
    gemm_bf16_wmma<false><<<dim3(DMODEL / 64, N_TOK / 128), 256, 0, stream>>>(
        attn_b, proj_wt, proj_b, (void*)out, N_TOK, DMODEL, DMODEL);
}
